// TensorProductConvLayer_42588895707436
// MI455X (gfx1250) — matrix operand order repaired and checked by reading, not compiled, not run on hardware
//
#include <hip/hip_runtime.h>

// ---- CDNA5 WMMA vector types ----
typedef __attribute__((ext_vector_type(16))) _Float16 v16h;
typedef __attribute__((ext_vector_type(8)))  _Float16 v8h;
typedef __attribute__((ext_vector_type(8)))  float    v8f;
typedef int v4i __attribute__((vector_size(16)));

#define NODE_DIM 56          // 32 + 3*8
#define WNUMEL 1600          // 1024 + 256 + 64 + 256
#define NTILES_W2 100        // 1600 / 16
#define WAVES 8
#define BLK_EDGES 128        // WAVES * 16
#define CT 4                 // N-tiles per streamed chunk
#define NCHUNK 25            // 100 / 4
#define CHUNK_HALVES (CT * 2 * 32 * 16)   // 4096 halves = 8 KB

#if __has_builtin(__builtin_amdgcn_global_load_async_to_lds_b128) && \
    __has_builtin(__builtin_amdgcn_s_wait_asynccnt)
#define USE_ASYNC_LDS 1
#else
#define USE_ASYNC_LDS 0
#endif

#define AS1 __attribute__((address_space(1)))
#define AS3 __attribute__((address_space(3)))

#if USE_ASYNC_LDS
// Copy 32 bytes global -> LDS via the gfx1250 async-to-LDS path (tracked by ASYNCcnt).
// Builtin signature: void(int4 AS1* src, int4 AS3* dst, imm offset, imm cpol).
// Integer casts: global flat == AS1 numerically; flat-to-LDS uses addr[31:0] (aperture rule).
__device__ __forceinline__ void async_cp32(const _Float16* g, _Float16* l) {
    __builtin_amdgcn_global_load_async_to_lds_b128(
        (AS1 v4i*)(uintptr_t)g,
        (AS3 v4i*)(unsigned int)(uintptr_t)l, 0, 0);
    __builtin_amdgcn_global_load_async_to_lds_b128(
        (AS1 v4i*)(uintptr_t)(g + 8),
        (AS3 v4i*)(unsigned int)(uintptr_t)(l + 8), 0, 0);
}
#endif

// ---------------- prep: pack weight matrix into f16 B-fragment layout ----------------
// B fragment (16x16x32 f16 WMMA): lane holds column N = nt*16 + (lane&15);
// lanes 0-15 hold K = kc*32 + 0..15, lanes 16-31 hold K = kc*32 + 16..31,
// stored as 16 contiguous halves per lane.
__global__ void prep_wfrag_kernel(const float* __restrict__ W, _Float16* __restrict__ out,
                                  int nCols, int nTiles) {
    int idx = blockIdx.x * blockDim.x + threadIdx.x;
    int total = nTiles * 2 * 32;
    if (idx >= total) return;
    int lane = idx & 31;
    int f    = idx >> 5;
    int kc   = f & 1;
    int nt   = f >> 1;
    int colN  = nt * 16 + (lane & 15);
    int kbase = kc * 32 + ((lane & 16) ? 16 : 0);
    _Float16* o = out + (size_t)idx * 16;
#pragma unroll
    for (int t = 0; t < 16; ++t)
        o[t] = (_Float16)W[(size_t)(kbase + t) * nCols + colN];
}

__global__ void zero_kernel(float* __restrict__ out, int nOut, float* __restrict__ cnt, int nCnt) {
    int i = blockIdx.x * blockDim.x + threadIdx.x;
    if (i < nOut) out[i] = 0.0f;
    if (i < nCnt) cnt[i] = 0.0f;
}

__global__ void finalize_kernel(float* __restrict__ out, const float* __restrict__ cnt,
                                const float* __restrict__ node_attr, int nElem) {
    int i = blockIdx.x * blockDim.x + threadIdx.x;
    if (i >= nElem) return;
    int row = i / NODE_DIM;
    float c = fmaxf(cnt[row], 1.0f);
    out[i] = out[i] / c + node_attr[i];
}

// Load an A fragment (16x32 f16) from a row-major [16][64] f16 LDS tile.
__device__ __forceinline__ v16h load_afrag_lds(const _Float16* lds, int kc, int lane) {
    int row = lane & 15;
    int off = (lane & 16) ? 16 : 0;
    const _Float16* p = lds + row * 64 + kc * 32 + off;
    v8h lo = *(const v8h*)p;
    v8h hi = *(const v8h*)(p + 8);
    v16h a;
#pragma unroll
    for (int i = 0; i < 8; ++i) { a[i] = lo[i]; a[i + 8] = hi[i]; }
    return a;
}

// ---------------- fused kernel ----------------
// 8 waves / block; each wave owns 16 edges; W2 B-fragments streamed via LDS once per block.
__global__ __launch_bounds__(256)
void fused_tp_kernel(const float* __restrict__ node_attr,
                     const int*   __restrict__ edge_index,   // [2][nE]
                     const float* __restrict__ edge_attr,    // [nE][64]
                     const float* __restrict__ edge_sh,      // [nE][4]
                     const float* __restrict__ fc_b1,        // [64]
                     const float* __restrict__ fc_b2,        // [1600]
                     const _Float16* __restrict__ w1f,       // packed B frags of fc_w1 (4096 halves)
                     const _Float16* __restrict__ w2f,       // packed B frags of fc_w2 (102400 halves)
                     float* __restrict__ out_acc,            // [nNodes][56]
                     float* __restrict__ cnt,                // [nNodes]
                     int nE) {
    __shared__ __align__(32) _Float16 lEA[WAVES][1024];   // per-wave staged edge_attr (f16)
    __shared__ __align__(32) _Float16 lH [WAVES][1024];   // per-wave hidden acts (f16)
    __shared__ float lc1[WAVES][16 * 32];                 // alpha*sh0*x0[u]      (w1 path)
    __shared__ float lp2[WAVES][16 * 32];                 // alpha*x0[u]          (w2 path)
    __shared__ float lc3[WAVES][16 * 24];                 // alpha*sh0*x1[u][j]   (w3 path)
    __shared__ float lc4[WAVES][16 * 8];                  // alpha/sqrt3*dot[u]   (w4 path)
    __shared__ __align__(32) _Float16 lw1[4096];          // W1 B-frags (whole block)
    __shared__ __align__(32) _Float16 bbuf[2][CHUNK_HALVES]; // W2 B-frag double buffer

    const int tid  = threadIdx.x;
    const int lane = tid & 31;
    const int wid  = tid >> 5;
    const int col  = lane & 15;
    const int hi   = lane >> 4;
    const long tb  = (long)blockIdx.x * BLK_EDGES;
    const long web = tb + (long)wid * 16;               // this wave's edge base
    const float alpha = 0.15811388300841898f;           // 1/sqrt(40)

    _Float16* myEA = &lEA[wid][0];
    _Float16* myH  = &lH [wid][0];
    float* lc1w = &lc1[wid][0];
    float* lp2w = &lp2[wid][0];
    float* lc3w = &lc3[wid][0];
    float* lc4w = &lc4[wid][0];

    // ---- stage this wave's 16x64 edge_attr tile (f16) ----
#pragma unroll
    for (int i = 0; i < 8; ++i) {
        int flat = lane * 8 + i;          // float4 slot (16 per edge row)
        int er   = flat >> 4;
        int c4   = flat & 15;
        long ee  = web + er; if (ee >= nE) ee = nE - 1;
        float4 v = ((const float4*)edge_attr)[ee * 16 + c4];
        int h0 = flat * 4;
        myEA[h0 + 0] = (_Float16)v.x; myEA[h0 + 1] = (_Float16)v.y;
        myEA[h0 + 2] = (_Float16)v.z; myEA[h0 + 3] = (_Float16)v.w;
    }
    // ---- stage W1 fragments (whole block cooperatively; 4096 halves) ----
    *(v16h*)(&lw1[tid * 16]) = *(const v16h*)(w1f + (size_t)tid * 16);

    // ---- gather node rows (edge_dst) + sh; precompute CG coefficients ----
    if (lane < 16) {
        long e = web + lane; if (e >= nE) e = nE - 1;
        int dst = edge_index[nE + e];
        const float* xr  = node_attr + (size_t)dst * NODE_DIM;
        const float* shr = edge_sh + (size_t)e * 4;
        float sh0 = shr[0], s1 = shr[1], s2 = shr[2], s3 = shr[3];
#pragma unroll
        for (int u = 0; u < 32; ++u) {
            float x0 = xr[u];
            lc1w[lane * 32 + u] = alpha * sh0 * x0;
            lp2w[lane * 32 + u] = alpha * x0;
        }
#pragma unroll
        for (int u = 0; u < 8; ++u) {
            float a = xr[32 + u * 3 + 0];
            float b = xr[32 + u * 3 + 1];
            float c = xr[32 + u * 3 + 2];
            lc3w[lane * 24 + u * 3 + 0] = alpha * sh0 * a;
            lc3w[lane * 24 + u * 3 + 1] = alpha * sh0 * b;
            lc3w[lane * 24 + u * 3 + 2] = alpha * sh0 * c;
            lc4w[lane * 8 + u] = alpha * 0.5773502691896258f * (a * s1 + b * s2 + c * s3);
        }
    }
    __syncthreads();

    // ---- GEMM1: H = relu(EA @ W1 + b1) ----
    {
        v16h aE0 = load_afrag_lds(myEA, 0, lane);
        v16h aE1 = load_afrag_lds(myEA, 1, lane);
#pragma unroll
        for (int nt = 0; nt < 4; ++nt) {
            v16h b0 = *(const v16h*)(lw1 + ((size_t)(nt * 2 + 0) * 32 + lane) * 16);
            v16h b1 = *(const v16h*)(lw1 + ((size_t)(nt * 2 + 1) * 32 + lane) * 16);
            v8f c = {0.f,0.f,0.f,0.f,0.f,0.f,0.f,0.f};
            c = __builtin_amdgcn_wmma_f32_16x16x32_f16(false, aE0, false, b0, (short)0, c, false, false);
            c = __builtin_amdgcn_wmma_f32_16x16x32_f16(false, aE1, false, b1, (short)0, c, false, false);
            float bias = fc_b1[nt * 16 + col];
#pragma unroll
            for (int r = 0; r < 8; ++r) {
                float h = c[r] + bias;
                h = h > 0.0f ? h : 0.0f;
                myH[(r + 8 * hi) * 64 + nt * 16 + col] = (_Float16)h;
            }
        }
    }

    // ---- prefetch W2 chunk 0 into bbuf[0] ----
#if USE_ASYNC_LDS
    async_cp32(w2f + (size_t)tid * 16, &bbuf[0][tid * 16]);
    __builtin_amdgcn_s_wait_asynccnt(0);
#else
    *(v16h*)(&bbuf[0][tid * 16]) = *(const v16h*)(w2f + (size_t)tid * 16);
#endif
    __syncthreads();

    // ---- A fragments for GEMM2 ----
    v16h aH0 = load_afrag_lds(myH, 0, lane);
    v16h aH1 = load_afrag_lds(myH, 1, lane);

    v8f acc0a = {0.f,0.f,0.f,0.f,0.f,0.f,0.f,0.f};   // out0 cols 0..15
    v8f acc0b = {0.f,0.f,0.f,0.f,0.f,0.f,0.f,0.f};   // out0 cols 16..31
    float q2[8];                                      // w2 path (sh1 applied at end)
    float o3[8][3];                                   // w3 path
#pragma unroll
    for (int r = 0; r < 8; ++r) { q2[r] = 0.f; o3[r][0] = 0.f; o3[r][1] = 0.f; o3[r][2] = 0.f; }
    const int usel = (lane & 15) >> 3;

    // ---- GEMM2 streamed over 25 chunks of 4 N-tiles, fused tensor-product ----
#pragma unroll 1
    for (int c = 0; c < NCHUNK; ++c) {
        const _Float16* cb = &bbuf[c & 1][0];
#if USE_ASYNC_LDS
        if (c + 1 < NCHUNK)
            async_cp32(w2f + (size_t)(c + 1) * CHUNK_HALVES + (size_t)tid * 16,
                       &bbuf[(c + 1) & 1][tid * 16]);
#else
        v16h pre;
        if (c + 1 < NCHUNK)
            pre = *(const v16h*)(w2f + (size_t)(c + 1) * CHUNK_HALVES + (size_t)tid * 16);
#endif
#pragma unroll
        for (int i = 0; i < CT; ++i) {
            int nt = c * CT + i;
            v16h b0 = *(const v16h*)(cb + ((size_t)(i * 2 + 0) * 32 + lane) * 16);
            v16h b1 = *(const v16h*)(cb + ((size_t)(i * 2 + 1) * 32 + lane) * 16);
            v8f d = {0.f,0.f,0.f,0.f,0.f,0.f,0.f,0.f};
            d = __builtin_amdgcn_wmma_f32_16x16x32_f16(false, aH0, false, b0, (short)0, d, false, false);
            d = __builtin_amdgcn_wmma_f32_16x16x32_f16(false, aH1, false, b1, (short)0, d, false, false);
            float b2v = fc_b2[nt * 16 + col];
#pragma unroll
            for (int r = 0; r < 8; ++r) d[r] += b2v;   // w includes +fc_b2

            if (nt < 64) {                 // w1: k = u*32 + wo -> out0
                int u = nt >> 1;
                if (nt & 1) {
#pragma unroll
                    for (int r = 0; r < 8; ++r) acc0b[r] += lc1w[(r + 8 * hi) * 32 + u] * d[r];
                } else {
#pragma unroll
                    for (int r = 0; r < 8; ++r) acc0a[r] += lc1w[(r + 8 * hi) * 32 + u] * d[r];
                }
            } else if (nt < 80) {          // w2: k = 1024 + u*8 + wo -> q2
                int u = 2 * (nt - 64) + usel;
#pragma unroll
                for (int r = 0; r < 8; ++r) q2[r] += lp2w[(r + 8 * hi) * 32 + u] * d[r];
            } else if (nt < 84) {          // w3: k = 1280 + u*8 + wo -> o3
                int u = 2 * (nt - 80) + usel;
#pragma unroll
                for (int r = 0; r < 8; ++r) {
                    int base = (r + 8 * hi) * 24 + u * 3;
                    o3[r][0] += lc3w[base + 0] * d[r];
                    o3[r][1] += lc3w[base + 1] * d[r];
                    o3[r][2] += lc3w[base + 2] * d[r];
                }
            } else {                       // w4: k = 1344 + u*32 + wo -> out0
                int t = nt - 84;
                int u = t >> 1;
                if (t & 1) {
#pragma unroll
                    for (int r = 0; r < 8; ++r) acc0b[r] += lc4w[(r + 8 * hi) * 8 + u] * d[r];
                } else {
#pragma unroll
                    for (int r = 0; r < 8; ++r) acc0a[r] += lc4w[(r + 8 * hi) * 8 + u] * d[r];
                }
            }
        }
#if USE_ASYNC_LDS
        __builtin_amdgcn_s_wait_asynccnt(0);
        __syncthreads();
#else
        __syncthreads();
        if (c + 1 < NCHUNK)
            *(v16h*)(&bbuf[(c + 1) & 1][tid * 16]) = pre;
        __syncthreads();
#endif
    }

    // ---- reduce out1 partials across the u-pair lanes (lane ^ 8) ----
#pragma unroll
    for (int r = 0; r < 8; ++r) {
        q2[r]    += __shfl_xor(q2[r],    8, 32);
        o3[r][0] += __shfl_xor(o3[r][0], 8, 32);
        o3[r][1] += __shfl_xor(o3[r][1], 8, 32);
        o3[r][2] += __shfl_xor(o3[r][2], 8, 32);
    }

    // ---- scatter-add by edge_src ----
#pragma unroll
    for (int r = 0; r < 8; ++r) {
        long e = web + r + 8 * hi;
        if (e < nE) {
            int s = edge_index[e];
            float* orow = out_acc + (size_t)s * NODE_DIM;
            atomicAdd(orow + col,      acc0a[r]);
            atomicAdd(orow + 16 + col, acc0b[r]);
            if ((lane & 15) < 8) {
                int wo = lane & 7;
                float s1 = edge_sh[e * 4 + 1];
                float s2 = edge_sh[e * 4 + 2];
                float s3 = edge_sh[e * 4 + 3];
                atomicAdd(orow + 32 + wo * 3 + 0, q2[r] * s1 + o3[r][0]);
                atomicAdd(orow + 32 + wo * 3 + 1, q2[r] * s2 + o3[r][1]);
                atomicAdd(orow + 32 + wo * 3 + 2, q2[r] * s3 + o3[r][2]);
            }
        }
    }
    if (lane < 16) {
        long e = web + lane;
        if (e < nE) atomicAdd(cnt + edge_index[e], 1.0f);
    }
}

// ---------------- host launcher ----------------
extern "C" void kernel_launch(void* const* d_in, const int* in_sizes, int n_in,
                              void* d_out, int out_size, void* d_ws, size_t ws_size,
                              hipStream_t stream) {
    const float* node_attr = (const float*)d_in[0];
    const int*   edge_index= (const int*)  d_in[1];
    const float* edge_attr = (const float*)d_in[2];
    const float* edge_sh   = (const float*)d_in[3];
    const float* fc_w1     = (const float*)d_in[4];
    const float* fc_b1     = (const float*)d_in[5];
    const float* fc_w2     = (const float*)d_in[6];
    const float* fc_b2     = (const float*)d_in[7];

    int nNodes = in_sizes[0] / NODE_DIM;
    int nE     = in_sizes[1] / 2;
    float* out = (float*)d_out;

    char* ws = (char*)d_ws;
    float*     cnt = (float*)ws;                   // nNodes floats
    _Float16*  w1f = (_Float16*)(ws + 200704);     // 4096 halves (8 KB)
    _Float16*  w2f = (_Float16*)(ws + 208896);     // 102400 halves (200 KB)

    int nOut = nNodes * NODE_DIM;
    zero_kernel<<<(nOut + 255) / 256, 256, 0, stream>>>(out, nOut, cnt, nNodes);

    prep_wfrag_kernel<<<1, 256, 0, stream>>>(fc_w1, w1f, 64, 4);
    prep_wfrag_kernel<<<(NTILES_W2 * 2 * 32 + 255) / 256, 256, 0, stream>>>(fc_w2, w2f, WNUMEL, NTILES_W2);

    int blocks = (nE + BLK_EDGES - 1) / BLK_EDGES;
    fused_tp_kernel<<<blocks, 256, 0, stream>>>(node_attr, edge_index, edge_attr, edge_sh,
                                                fc_b1, fc_b2, w1f, w2f, out, cnt, nE);

    finalize_kernel<<<(nOut + 255) / 256, 256, 0, stream>>>(out, cnt, node_attr, nOut);
}
